// WMSA_66864050864840
// MI455X (gfx1250) — matrix operand order repaired and checked by reading, not compiled, not run on hardware
//
#include <hip/hip_runtime.h>
#include <hip/hip_bf16.h>

// ---------------------------------------------------------------------------
// Windowed MSA for MI455X (gfx1250), wave32, WMMA f16->f32.
// Sizes: B=32, C=384, H=W=64, M=8 (strided windows), NH=12, HD=32.
// Tokens = 131072; windows = 2048; tokens/window = 64.
// ---------------------------------------------------------------------------

typedef __attribute__((ext_vector_type(16))) _Float16     v16h;
typedef __attribute__((ext_vector_type(8)))  float        v8f;
typedef __attribute__((ext_vector_type(4)))  unsigned int u32x4;

#define WMMA(a, b, c) \
  __builtin_amdgcn_wmma_f32_16x16x32_f16(false, (a), false, (b), (short)0, (c), false, false)

// ---- fragment loaders (layouts per CDNA5 ISA 7.12.2, wave32) ---------------
// A 16x32 f16: lanes 0-15 -> M=lane, K = {0..7, 16..23}; lanes 16-31 -> K = {8..15, 24..31}
__device__ __forceinline__ v16h frag_a16(const _Float16* p, int row, int stride, int lane) {
  int kh = (lane < 16) ? 0 : 8;
  const _Float16* q = p + row * stride + kh;
  v16h a;
  ((u32x4*)&a)[0] = *(const u32x4*)(q);
  ((u32x4*)&a)[1] = *(const u32x4*)(q + 16);
  return a;
}
// B 32x16 f16 stored in LDS as [n][k] (k contiguous): lanes 0-15 -> K={0..7,16..23},
// lanes 16-31 -> K={8..15,24..31}, N = lane&15.  Two 8-half (16B) chunks.
__device__ __forceinline__ v16h frag_b16(const _Float16* p, int n, int stride, int kofs, int lane) {
  int kb = kofs + ((lane < 16) ? 0 : 8);
  const _Float16* q = p + n * stride + kb;
  v16h b;
  ((u32x4*)&b)[0] = *(const u32x4*)(q);
  ((u32x4*)&b)[1] = *(const u32x4*)(q + 16);
  return b;
}

// ---------------------------------------------------------------------------
// Kernel 1: gather x [B,C,64,64] f32 -> xh [win][tok][C] f16  (token-major)
// Window mapping: pixel (r,cc): win = (b*8 + r%8)*8 + cc%8, tok = (r/8)*8 + cc/8
// ---------------------------------------------------------------------------
__global__ __launch_bounds__(256) void pack_x(const float* __restrict__ x,
                                              _Float16* __restrict__ xh) {
  __shared__ _Float16 tile[64 * 72]; // [col][c], padded
  int r   = blockIdx.x;              // row 0..63
  int bg  = blockIdx.y;              // 0..B*6-1
  int b   = bg / 6, c0 = (bg % 6) * 64;
  int tid = threadIdx.x;
  const float* src = x + (((size_t)(b * 384 + c0)) * 64 + r) * 64;
  for (int it = 0; it < 16; ++it) {
    int idx = tid + it * 256;
    int c = idx >> 6, col = idx & 63;                   // coalesced global read
    tile[col * 72 + c] = (_Float16)src[(size_t)c * 4096 + col];
  }
  __syncthreads();
  int m1 = r >> 3, hp = r & 7;
  for (int it = 0; it < 16; ++it) {
    int idx = tid + it * 256;
    int col = idx >> 6, c = idx & 63;                   // coalesced global write
    int m2 = col >> 3, wp = col & 7;
    int win = (b * 8 + hp) * 8 + wp;
    int t   = m1 * 8 + m2;
    xh[((size_t)(win * 64 + t)) * 384 + c0 + c] = tile[col * 72 + c];
  }
}

// ---------------------------------------------------------------------------
// Kernel 2: fused QKV projection.  Y = Xh @ W^T for W in {Wq,Wk,Wv}.
// Block = 256 threads (8 waves), tile 128(M) x 64(N), K-step 32.
// ---------------------------------------------------------------------------
__global__ __launch_bounds__(256) void gemm_qkv(
    const _Float16* __restrict__ xh,
    const float* __restrict__ Wq, const float* __restrict__ Wk, const float* __restrict__ Wv,
    _Float16* __restrict__ qh, _Float16* __restrict__ kh, _Float16* __restrict__ vh) {
  __shared__ _Float16 As[128 * 40];       // [m][k], k contiguous, 16B-aligned rows
  __shared__ _Float16 Bs[3][64 * 40];     // [n][k]
  int tid  = threadIdx.x;
  int wid  = tid >> 5, lane = tid & 31;
  int m0   = blockIdx.x * 128, n0 = blockIdx.y * 64;
  int mi   = wid & 3, ni = wid >> 2;
  const float* Ws[3] = {Wq, Wk, Wv};
  v8f acc[3][2][2] = {};
  for (int k0 = 0; k0 < 384; k0 += 32) {
    __syncthreads();
    for (int it = 0; it < 2; ++it) {                    // A: 128x32 halves
      int idx = tid + it * 256;
      int row = idx >> 2, kq = (idx & 3) << 3;
      *(u32x4*)&As[row * 40 + kq] = *(const u32x4*)(xh + (size_t)(m0 + row) * 384 + k0 + kq);
    }
    {                                                   // B: 3 x 64x32, f32 -> f16
      int row = tid >> 2, kq = (tid & 3) << 3;
      for (int mtx = 0; mtx < 3; ++mtx) {
        const float* wp = Ws[mtx] + (size_t)(n0 + row) * 384 + k0 + kq;
        _Float16* bp = &Bs[mtx][row * 40 + kq];
#pragma unroll
        for (int u = 0; u < 8; ++u) bp[u] = (_Float16)wp[u];
      }
    }
    __syncthreads();
    v16h a[2], b[3][2];
#pragma unroll
    for (int fm = 0; fm < 2; ++fm)
      a[fm] = frag_a16(As, mi * 32 + fm * 16 + (lane & 15), 40, lane);
#pragma unroll
    for (int mtx = 0; mtx < 3; ++mtx)
#pragma unroll
      for (int fn = 0; fn < 2; ++fn)
        b[mtx][fn] = frag_b16(Bs[mtx], ni * 32 + fn * 16 + (lane & 15), 40, 0, lane);
#pragma unroll
    for (int mtx = 0; mtx < 3; ++mtx)
#pragma unroll
      for (int fm = 0; fm < 2; ++fm)
#pragma unroll
        for (int fn = 0; fn < 2; ++fn)
          acc[mtx][fm][fn] = WMMA(a[fm], b[mtx][fn], acc[mtx][fm][fn]);
  }
  _Float16* outs[3] = {qh, kh, vh};
  int lhalf = (lane < 16) ? 0 : 8, ncol = lane & 15;
  for (int mtx = 0; mtx < 3; ++mtx)
    for (int fm = 0; fm < 2; ++fm)
      for (int fn = 0; fn < 2; ++fn)
#pragma unroll
        for (int r = 0; r < 8; ++r) {
          int gm = m0 + mi * 32 + fm * 16 + lhalf + r;
          int gc = n0 + ni * 32 + fn * 16 + ncol;
          outs[mtx][(size_t)gm * 384 + gc] = (_Float16)acc[mtx][fm][fn][r];
        }
}

// ---------------------------------------------------------------------------
// Kernel 3: attention per (window, head).  Block = 128 threads (4 waves).
// S = Q K^T (WMMA) -> +bias, softmax in LDS -> write attn_weight (f32)
// O = P V (WMMA) -> oh f16.
// ---------------------------------------------------------------------------
__global__ __launch_bounds__(128) void attn_kernel(
    const _Float16* __restrict__ qh, const _Float16* __restrict__ kh,
    const _Float16* __restrict__ vh, const float* __restrict__ bias_mat,
    float* __restrict__ aw, _Float16* __restrict__ oh) {
  __shared__ _Float16 Qs[64 * 40];   // [i][c]  (A layout)
  __shared__ _Float16 Ks[64 * 40];   // [j][c]  (B layout for S: n=j, k=c)
  __shared__ _Float16 Vt[32 * 72];   // [c][j]  (B layout for O: n=c, k=j)
  __shared__ float    Sf[64 * 65];   // scores/probs, padded stride 65
  __shared__ float    bm[225];
  int tid  = threadIdx.x;
  int win  = blockIdx.x / 12, head = blockIdx.x - win * 12;
  size_t base = (size_t)win * 64 * 384 + (size_t)head * 32;
  const _Float16* qg = qh + base;
  const _Float16* kg = kh + base;
  const _Float16* vg = vh + base;
  for (int it = 0; it < 2; ++it) {
    int idx = tid + it * 128;
    int row = idx >> 2, kq = (idx & 3) << 3;
    *(u32x4*)&Qs[row * 40 + kq] = *(const u32x4*)(qg + (size_t)row * 384 + kq);
    *(u32x4*)&Ks[row * 40 + kq] = *(const u32x4*)(kg + (size_t)row * 384 + kq);
  }
  for (int it = 0; it < 16; ++it) {
    int idx = tid + it * 128;
    int j = idx >> 5, c = idx & 31;
    Vt[c * 72 + j] = vg[(size_t)j * 384 + c];
  }
  for (int i = tid; i < 225; i += 128) bm[i] = bias_mat[i];
  __syncthreads();

  int wid = tid >> 5, lane = tid & 31;
  int mbase = wid * 16;
  int lrow = lane & 15, lhalf = (lane < 16) ? 0 : 8;
  {   // S = Q K^T : each wave does 16 rows x 64 cols (4 WMMAs)
    v16h a = frag_a16(Qs, mbase + lrow, 40, lane);
    v8f  s[4] = {};
#pragma unroll
    for (int nt = 0; nt < 4; ++nt) {
      v16h b = frag_b16(Ks, nt * 16 + lrow, 40, 0, lane);
      s[nt] = WMMA(a, b, s[nt]);
    }
#pragma unroll
    for (int nt = 0; nt < 4; ++nt)
#pragma unroll
      for (int r = 0; r < 8; ++r)
        Sf[(mbase + lhalf + r) * 65 + nt * 16 + lrow] = s[nt][r];
  }
  __syncthreads();

  if (tid < 64) {  // scale + rel-pos bias + softmax, one thread per row i
    int i = tid, x1 = i >> 3, y1 = i & 7;
    const float scale = 0.05103103630798288f;   // 384^-0.5 (hidden_dim, per source)
    float mx = -3.4e38f;
    for (int j = 0; j < 64; ++j) {
      int x2 = j >> 3, y2 = j & 7;
      // rel = pos[j] - pos[i]  (mod 15)
      float s = Sf[i * 65 + j] * scale +
                bm[((x2 - x1 + 15) % 15) * 15 + ((y2 - y1 + 15) % 15)];
      Sf[i * 65 + j] = s;
      mx = fmaxf(mx, s);
    }
    float sum = 0.f;
    for (int j = 0; j < 64; ++j) {
      float e = __expf(Sf[i * 65 + j] - mx);
      Sf[i * 65 + j] = e;
      sum += e;
    }
    float inv = 1.f / sum;
    float* awp = aw + ((size_t)(win * 12 + head) * 64 + i) * 64;
    for (int j = 0; j < 64; ++j) {
      float p = Sf[i * 65 + j] * inv;
      Sf[i * 65 + j] = p;
      awp[j] = p;
    }
  }
  __syncthreads();

  // O = P V : wave does 16 rows x 32 cols, K=64 (2 k-steps x 2 n-tiles)
  v8f o[2] = {};
#pragma unroll
  for (int ks = 0; ks < 2; ++ks) {
    int jb = ks * 32;
    v16h pa;
    int prow = mbase + lrow;
#pragma unroll
    for (int u = 0; u < 8; ++u) pa[u]     = (_Float16)Sf[prow * 65 + jb + lhalf + u];
#pragma unroll
    for (int u = 0; u < 8; ++u) pa[u + 8] = (_Float16)Sf[prow * 65 + jb + lhalf + 16 + u];
#pragma unroll
    for (int ct = 0; ct < 2; ++ct) {
      v16h vb = frag_b16(Vt, ct * 16 + lrow, 72, jb, lane);
      o[ct] = WMMA(pa, vb, o[ct]);
    }
  }
  _Float16* og = oh + base;
#pragma unroll
  for (int ct = 0; ct < 2; ++ct)
#pragma unroll
    for (int r = 0; r < 8; ++r)
      og[(size_t)(mbase + lhalf + r) * 384 + ct * 16 + lrow] = (_Float16)o[ct][r];
}

// ---------------------------------------------------------------------------
// Kernel 4: out = (O @ Wo^T) scattered back to [B,C,64,64] f32.
// ---------------------------------------------------------------------------
__global__ __launch_bounds__(256) void gemm_out(
    const _Float16* __restrict__ oh, const float* __restrict__ Wo,
    float* __restrict__ out) {
  __shared__ _Float16 As[128 * 40];
  __shared__ _Float16 Bs[64 * 40];
  int tid = threadIdx.x;
  int wid = tid >> 5, lane = tid & 31;
  int m0 = blockIdx.x * 128, n0 = blockIdx.y * 64;
  int mi = wid & 3, ni = wid >> 2;
  v8f acc[2][2] = {};
  for (int k0 = 0; k0 < 384; k0 += 32) {
    __syncthreads();
    for (int it = 0; it < 2; ++it) {
      int idx = tid + it * 256;
      int row = idx >> 2, kq = (idx & 3) << 3;
      *(u32x4*)&As[row * 40 + kq] = *(const u32x4*)(oh + (size_t)(m0 + row) * 384 + k0 + kq);
    }
    {
      int row = tid >> 2, kq = (tid & 3) << 3;
      const float* wp = Wo + (size_t)(n0 + row) * 384 + k0 + kq;
      _Float16* bp = &Bs[row * 40 + kq];
#pragma unroll
      for (int u = 0; u < 8; ++u) bp[u] = (_Float16)wp[u];
    }
    __syncthreads();
    v16h a[2], bf[2];
#pragma unroll
    for (int fm = 0; fm < 2; ++fm)
      a[fm] = frag_a16(As, mi * 32 + fm * 16 + (lane & 15), 40, lane);
#pragma unroll
    for (int fn = 0; fn < 2; ++fn)
      bf[fn] = frag_b16(Bs, ni * 32 + fn * 16 + (lane & 15), 40, 0, lane);
#pragma unroll
    for (int fm = 0; fm < 2; ++fm)
#pragma unroll
      for (int fn = 0; fn < 2; ++fn)
        acc[fm][fn] = WMMA(a[fm], bf[fn], acc[fm][fn]);
  }
  int lhalf = (lane < 16) ? 0 : 8, ncol = lane & 15;
  for (int fm = 0; fm < 2; ++fm)
    for (int fn = 0; fn < 2; ++fn)
#pragma unroll
      for (int r = 0; r < 8; ++r) {
        int gm = m0 + mi * 32 + fm * 16 + lhalf + r;    // token
        int gc = n0 + ni * 32 + fn * 16 + ncol;         // out channel
        int wn = gm >> 6, t = gm & 63;
        int bb = wn >> 6, hp = (wn >> 3) & 7, wp = wn & 7;
        int rr = (t >> 3) * 8 + hp, cc = (t & 7) * 8 + wp;
        out[(size_t)bb * 1572864 + (size_t)gc * 4096 + rr * 64 + cc] = acc[fm][fn][r];
      }
}

// ---------------------------------------------------------------------------
extern "C" void kernel_launch(void* const* d_in, const int* in_sizes, int n_in,
                              void* d_out, int out_size, void* d_ws, size_t ws_size,
                              hipStream_t stream) {
  (void)in_sizes; (void)n_in; (void)out_size; (void)ws_size;
  const float* x        = (const float*)d_in[0];
  const float* Wq       = (const float*)d_in[1];
  const float* Wk       = (const float*)d_in[2];
  const float* Wv       = (const float*)d_in[3];
  const float* Wo       = (const float*)d_in[4];
  const float* bias_mat = (const float*)d_in[5];

  float* out = (float*)d_out;                 // [32,384,64,64]
  float* aw  = out + 50331648;                // attn_weight [2048,12,64,64] f32

  const size_t SZB = 100663296;               // 131072*384 halves in bytes
  char* ws = (char*)d_ws;
  _Float16* xh = (_Float16*)(ws);             // token-major activations
  _Float16* qh = (_Float16*)(ws + 1 * SZB);
  _Float16* kh = (_Float16*)(ws + 2 * SZB);
  _Float16* vh = (_Float16*)(ws + 3 * SZB);
  _Float16* oh = xh;                          // reuse x buffer once consumed

  pack_x   <<<dim3(64, 192), 256, 0, stream>>>(x, xh);
  gemm_qkv <<<dim3(1024, 6), 256, 0, stream>>>(xh, Wq, Wk, Wv, qh, kh, vh);
  attn_kernel<<<dim3(24576), 128, 0, stream>>>(qh, kh, vh, bias_mat, aw, oh);
  gemm_out <<<dim3(1024, 6), 256, 0, stream>>>(oh, Wo, out);
}
